// MultiHeadAttention_33483565039672
// MI455X (gfx1250) — matrix operand order repaired and checked by reading, not compiled, not run on hardware
//
#include <hip/hip_runtime.h>
#include <hip/hip_bf16.h>
#include <stdint.h>

// ---------------------------------------------------------------------------
// MI455X (gfx1250) fused MHA + LayerNorm, wave32 + v_wmma_f32_16x16x32_f16.
//   B=2, S=2048, D=1024, H=16, DK=DV=64
// Pipeline:
//   0) mask -> bitmask pack (wave32 __ballot), 134MB of repeat traffic -> 1MB
//   1..3) QKV projections: f16-WMMA GEMM (256x64 block tile, K-step 64,
//         register-prefetched staging), epilogue scatters into reshape-based
//         head layout; V stored transposed [B,H,64,S]
//   4) flash attention: double-buffered async-to-LDS K/V tiles, online
//      softmax fp32, P@V via WMMA, context -> [B,S,H*DV] f16
//   5) output projection + bias + residual (fp32)
//   6) LayerNorm -> d_out (fp32)
// Workspace: 4 x 8MB f16 + 16MB fp32 + 1MB bitmask  (~49MB)
// ---------------------------------------------------------------------------

typedef _Float16 half_t;
typedef __attribute__((ext_vector_type(8)))  _Float16 v8h;
typedef __attribute__((ext_vector_type(16))) _Float16 v16h;
typedef __attribute__((ext_vector_type(8)))  float    v8f;
typedef __attribute__((ext_vector_type(4)))  float    f32x4;

#define CB 2
#define CS 2048
#define CD 1024
#define CH 16
#define CDK 64

__device__ __forceinline__ v8f zero8() {
    v8f z;
#pragma unroll
    for (int i = 0; i < 8; ++i) z[i] = 0.0f;
    return z;
}

__device__ __forceinline__ v16h cat16(v8h lo, v8h hi) {
    v16h r;
#pragma unroll
    for (int i = 0; i < 8; ++i) { r[i] = lo[i]; r[i + 8] = hi[i]; }
    return r;
}

__device__ __forceinline__ v8f wmma_f16(v16h a, v16h b, v8f c) {
    return __builtin_amdgcn_wmma_f32_16x16x32_f16(
        false, a, false, b, (short)0, c, false, false);
}

// A-fragment (16x32 f16): lane<16 holds K {0..7,16..23}; lane>=16 {8..15,24..31}
__device__ __forceinline__ v16h load_afrag(const half_t* row, int hiHalf) {
    v8h lo = *(const v8h*)(row + hiHalf * 8);
    v8h hi = *(const v8h*)(row + hiHalf * 8 + 16);
    return cat16(lo, hi);
}

// B-fragment (32x16 f16): lane<16 holds K {0..7,16..23} of its column; lane>=16 {8..15,24..31}
__device__ __forceinline__ v16h load_bfrag(const half_t* row, int hiHalf) {
    v8h lo = *(const v8h*)(row + hiHalf * 8);
    v8h hi = *(const v8h*)(row + hiHalf * 8 + 16);
    return cat16(lo, hi);
}

// ---------------------------------------------------------------------------
// Mask bit-pack: bits[row][w] bit L = (mask[row][w*32+L] != 0)
// ---------------------------------------------------------------------------
__global__ __launch_bounds__(256) void mask_pack_kernel(
    const unsigned char* __restrict__ mask, uint32_t* __restrict__ bits)
{
    const int row  = blockIdx.x;
    const int wv   = threadIdx.x >> 5;
    const int lane = threadIdx.x & 31;
    const unsigned char* mrow = mask + (size_t)row * CS;
#pragma unroll
    for (int i = 0; i < 8; ++i) {
        const int word = i * 8 + wv;
        const int pred = mrow[word * 32 + lane] != 0;
        unsigned long long bal = __ballot(pred);
        if (lane == 0) bits[(size_t)row * (CS / 32) + word] = (uint32_t)bal;
    }
}

// ---------------------------------------------------------------------------
// GEMM: out[r][c] = sum_k A[r][k] * W[c][k] + bias[c]   (M=4096, N=1024, K=1024)
// Block tile 256x64, wave tile 64x32, K-step 64, register-prefetched staging.
// MODE 0: f16 -> q/k head layout [B,H,S,64]
// MODE 1: f16 -> transposed v layout [B,H,64,S]
// MODE 2: fp32 -> acc + bias + residual
// ---------------------------------------------------------------------------
template <typename AT, int MODE>
__global__ __launch_bounds__(256) void gemm_kernel(
    const AT* __restrict__ A, const float* __restrict__ W,
    const float* __restrict__ bias, const float* __restrict__ resid,
    half_t* __restrict__ outH, float* __restrict__ outF)
{
    __shared__ __align__(16) half_t sA[256][72];
    __shared__ __align__(16) half_t sB[64][72];

    const int tid  = threadIdx.x;
    const int wv   = tid >> 5;
    const int lane = tid & 31;
    const int lh   = lane & 15;
    const int hiH  = lane >> 4;
    const int wm   = wv & 3;           // 4 M-subtiles of 64
    const int wn   = wv >> 2;          // 2 N-subtiles of 32
    const int m0   = blockIdx.y * 256;
    const int n0   = blockIdx.x * 64;

    v8f acc[4][2];
#pragma unroll
    for (int mi = 0; mi < 4; ++mi)
#pragma unroll
        for (int ni = 0; ni < 2; ++ni) acc[mi][ni] = zero8();

    // staging maps: A row = tid (64 k each); B row = tid>>2, 16 k each
    const int brr = tid >> 2, bkk = (tid & 3) * 16;

    f32x4 pa[16]; f32x4 pb[4];   // fp32 prefetch regs
    v8h   pah[8]; v8h   pbh[2];  // f16 prefetch regs (AT == half path uses pah)

    // ---- prefetch kk = 0 ----
    {
        if constexpr (__is_same(AT, float)) {
            const float* srcA = A + (size_t)(m0 + tid) * CD;
#pragma unroll
            for (int i = 0; i < 16; ++i) pa[i] = *(const f32x4*)(srcA + i * 4);
        } else {
            const half_t* srcA = A + (size_t)(m0 + tid) * CD;
#pragma unroll
            for (int i = 0; i < 8; ++i) pah[i] = *(const v8h*)(srcA + i * 8);
        }
        const float* srcB = W + (size_t)(n0 + brr) * CD + bkk;
#pragma unroll
        for (int i = 0; i < 4; ++i) pb[i] = *(const f32x4*)(srcB + i * 4);
    }

    for (int kk = 0; kk < CD; kk += 64) {
        __syncthreads();   // previous compute done with LDS
        // ---- store staged tile ----
        if constexpr (__is_same(AT, float)) {
#pragma unroll
            for (int i = 0; i < 8; ++i) {
                v8h h;
#pragma unroll
                for (int e = 0; e < 4; ++e) h[e]     = (half_t)pa[2 * i][e];
#pragma unroll
                for (int e = 0; e < 4; ++e) h[4 + e] = (half_t)pa[2 * i + 1][e];
                *(v8h*)&sA[tid][i * 8] = h;
            }
        } else {
#pragma unroll
            for (int i = 0; i < 8; ++i) *(v8h*)&sA[tid][i * 8] = pah[i];
        }
#pragma unroll
        for (int i = 0; i < 2; ++i) {
            v8h h;
#pragma unroll
            for (int e = 0; e < 4; ++e) h[e]     = (half_t)pb[2 * i][e];
#pragma unroll
            for (int e = 0; e < 4; ++e) h[4 + e] = (half_t)pb[2 * i + 1][e];
            *(v8h*)&sB[brr][bkk + i * 8] = h;
        }
        __syncthreads();

        // ---- prefetch next tile (overlaps with WMMA below) ----
        if (kk + 64 < CD) {
            if constexpr (__is_same(AT, float)) {
                const float* srcA = A + (size_t)(m0 + tid) * CD + kk + 64;
#pragma unroll
                for (int i = 0; i < 16; ++i) pa[i] = *(const f32x4*)(srcA + i * 4);
            } else {
                const half_t* srcA = A + (size_t)(m0 + tid) * CD + kk + 64;
#pragma unroll
                for (int i = 0; i < 8; ++i) pah[i] = *(const v8h*)(srcA + i * 8);
            }
            const float* srcB = W + (size_t)(n0 + brr) * CD + kk + 64 + bkk;
#pragma unroll
            for (int i = 0; i < 4; ++i) pb[i] = *(const f32x4*)(srcB + i * 4);
        }

        // ---- compute: 2 K-chunks x (4 x 2) WMMA ----
#pragma unroll
        for (int kc = 0; kc < 2; ++kc) {
            v16h aF[4], bF[2];
#pragma unroll
            for (int mi = 0; mi < 4; ++mi)
                aF[mi] = load_afrag(&sA[wm * 64 + mi * 16 + lh][kc * 32], hiH);
#pragma unroll
            for (int ni = 0; ni < 2; ++ni)
                bF[ni] = load_bfrag(&sB[wn * 32 + ni * 16 + lh][kc * 32], hiH);
#pragma unroll
            for (int mi = 0; mi < 4; ++mi)
#pragma unroll
                for (int ni = 0; ni < 2; ++ni)
                    acc[mi][ni] = wmma_f16(aF[mi], bF[ni], acc[mi][ni]);
        }
    }

    // ---- epilogue ----
#pragma unroll
    for (int mi = 0; mi < 4; ++mi) {
#pragma unroll
        for (int ni = 0; ni < 2; ++ni) {
            const int c = n0 + wn * 32 + ni * 16 + lh;
#pragma unroll
            for (int j = 0; j < 8; ++j) {
                const int r = m0 + wm * 64 + mi * 16 + hiH * 8 + j;
                float v = acc[mi][ni][j] + bias[c];
                if constexpr (MODE == 2) {
                    const size_t idx = (size_t)r * CD + c;
                    outF[idx] = v + resid[idx];
                } else {
                    const int b   = r >> 11;
                    const int s   = r & 2047;
                    const int h   = s >> 7;
                    const int pos = ((s & 127) << 4) | (c >> 6);
                    const int d   = c & 63;
                    if constexpr (MODE == 0) {
                        outH[((((size_t)b * CH + h) * CS + pos) << 6) + d] = (half_t)v;
                    } else { // MODE 1: transposed V
                        outH[(((size_t)b * CH + h) * CDK + d) * CS + pos] = (half_t)v;
                    }
                }
            }
        }
    }
}

// ---------------------------------------------------------------------------
// Flash attention: block = (b, h, 128 q rows); 8 waves x 16 rows.
// Double-buffered async-to-LDS K/V tiles of 32 keys; bitmask masking.
// ---------------------------------------------------------------------------
__global__ __launch_bounds__(256) void flash_attn_kernel(
    const half_t* __restrict__ qh, const half_t* __restrict__ kh,
    const half_t* __restrict__ vT, const uint32_t* __restrict__ mbits,
    half_t* __restrict__ ctx)
{
    __shared__ __align__(16) half_t sK[2][32][72];    // [buf][key][d]
    __shared__ __align__(16) half_t sV[2][64][40];    // [buf][d][key]
    __shared__ __align__(16) half_t sP[8][16][40];    // per-wave P tile

    const int tid  = threadIdx.x;
    const int wv   = tid >> 5;
    const int lane = tid & 31;
    const int lh   = lane & 15;
    const int hiH  = lane >> 4;
    const int b    = blockIdx.z, h = blockIdx.y;
    const int q0   = blockIdx.x * 128 + wv * 16;

    const half_t* Qb = qh + (((size_t)b * CH + h) * CS) * CDK;
    const half_t* Kb = kh + (((size_t)b * CH + h) * CS) * CDK;
    const half_t* Vb = vT + (((size_t)b * CH + h) * CDK) * CS;
    const uint32_t* Mb = mbits + (size_t)b * CS * (CS / 32);

    // Q fragments (loop invariant)
    v16h qF[2];
#pragma unroll
    for (int kc = 0; kc < 2; ++kc) {
        const half_t* qr = Qb + (size_t)(q0 + lh) * CDK + kc * 32 + hiH * 8;
        qF[kc] = cat16(*(const v8h*)qr, *(const v8h*)(qr + 16));
    }

    v8f accO[4];
#pragma unroll
    for (int f = 0; f < 4; ++f) accO[f] = zero8();
    float mrow[8], lrow[8];
#pragma unroll
    for (int j = 0; j < 8; ++j) { mrow[j] = -1e30f; lrow[j] = 0.0f; }

    // async staging: per-thread 16B of K tile and V tile
    const int kr = tid >> 3, kko = (tid & 7) * 8;   // K tile: 32 x 64 halves
    const int vr = tid >> 2, vko = (tid & 3) * 8;   // V tile: 64 x 32 halves
    uint32_t ldsK[2], ldsV[2];
#pragma unroll
    for (int p = 0; p < 2; ++p) {
        ldsK[p] = (uint32_t)(uintptr_t)&sK[p][kr][kko];
        ldsV[p] = (uint32_t)(uintptr_t)&sV[p][vr][vko];
    }
    const half_t* gK0 = Kb + (size_t)kr * CDK + kko;
    const half_t* gV0 = Vb + (size_t)vr * CS + vko;

#define ISSUE_TILE(kt0_, buf_)                                                  \
    do {                                                                        \
        uint64_t ga_ = (uint64_t)(uintptr_t)(gK0 + (size_t)(kt0_) * CDK);       \
        uint64_t gv_ = (uint64_t)(uintptr_t)(gV0 + (kt0_));                     \
        asm volatile("global_load_async_to_lds_b128 %0, %1, off"                \
                     :: "v"(ldsK[buf_]), "v"(ga_) : "memory");                  \
        asm volatile("global_load_async_to_lds_b128 %0, %1, off"                \
                     :: "v"(ldsV[buf_]), "v"(gv_) : "memory");                  \
    } while (0)

    ISSUE_TILE(0, 0);
    asm volatile("s_wait_asynccnt 0" ::: "memory");
    __syncthreads();

    for (int kt = 0; kt < CS / 32; ++kt) {
        const int cur = kt & 1;
        if (kt + 1 < CS / 32) ISSUE_TILE((kt + 1) * 32, cur ^ 1);

        // scores: 16x32 per wave = 2 N-tiles, K-depth 64 = 2 wmma each
        v8f sc[2];
#pragma unroll
        for (int ni = 0; ni < 2; ++ni) {
            v8f a = zero8();
#pragma unroll
            for (int kc = 0; kc < 2; ++kc) {
                v16h bK = load_bfrag(&sK[cur][ni * 16 + lh][kc * 32], hiH);
                a = wmma_f16(qF[kc], bK, a);
            }
            sc[ni] = a;
        }

        // online softmax (fp32); row j in vgpr j, lane-half selects +8
        float p0[8], p1[8], scl[8];
#pragma unroll
        for (int j = 0; j < 8; ++j) {
            float s0 = sc[0][j] * 0.125f;
            float s1 = sc[1][j] * 0.125f;
            const uint32_t mw =
                Mb[(size_t)(q0 + hiH * 8 + j) * (CS / 32) + kt];
            if ((mw >> lh) & 1u)        s0 = -1e9f;
            if ((mw >> (lh + 16)) & 1u) s1 = -1e9f;
            float t = fmaxf(s0, s1);
            t = fmaxf(t, __shfl_xor(t, 1));
            t = fmaxf(t, __shfl_xor(t, 2));
            t = fmaxf(t, __shfl_xor(t, 4));
            t = fmaxf(t, __shfl_xor(t, 8));
            const float mn = fmaxf(mrow[j], t);
            const float sf = __expf(mrow[j] - mn);
            p0[j] = __expf(s0 - mn);
            p1[j] = __expf(s1 - mn);
            float rs = p0[j] + p1[j];
            rs += __shfl_xor(rs, 1);
            rs += __shfl_xor(rs, 2);
            rs += __shfl_xor(rs, 4);
            rs += __shfl_xor(rs, 8);
            lrow[j] = lrow[j] * sf + rs;
            mrow[j] = mn;
            scl[j]  = sf;
        }
#pragma unroll
        for (int f = 0; f < 4; ++f)
#pragma unroll
            for (int j = 0; j < 8; ++j) accO[f][j] *= scl[j];

        // P (C-layout) -> LDS -> A-fragment; wave-private, LDS in-order per wave
#pragma unroll
        for (int j = 0; j < 8; ++j) {
            sP[wv][hiH * 8 + j][lh]      = (half_t)p0[j];
            sP[wv][hiH * 8 + j][16 + lh] = (half_t)p1[j];
        }
        asm volatile("" ::: "memory");
        v16h pF = load_afrag(&sP[wv][lh][0], hiH);

        // O += P @ V : 4 N-tiles over DV=64
#pragma unroll
        for (int f = 0; f < 4; ++f) {
            v16h bV = load_bfrag(&sV[cur][f * 16 + lh][0], hiH);
            accO[f] = wmma_f16(pF, bV, accO[f]);
        }

        asm volatile("s_wait_asynccnt 0" ::: "memory");
        __syncthreads();
    }
#undef ISSUE_TILE

    // normalize + store context [B,S,H*DV] f16
    float inv[8];
#pragma unroll
    for (int j = 0; j < 8; ++j) inv[j] = 1.0f / lrow[j];
#pragma unroll
    for (int f = 0; f < 4; ++f) {
#pragma unroll
        for (int j = 0; j < 8; ++j) {
            const int row = q0 + hiH * 8 + j;
            ctx[((size_t)b * CS + row) * (CH * CDK) + h * CDK + f * 16 + lh] =
                (half_t)(accO[f][j] * inv[j]);
        }
    }
}

// ---------------------------------------------------------------------------
// LayerNorm over D=1024 per row
// ---------------------------------------------------------------------------
__global__ __launch_bounds__(256) void layernorm_kernel(
    const float* __restrict__ x, const float* __restrict__ gamma,
    const float* __restrict__ beta, float* __restrict__ out)
{
    __shared__ float rsum[8], rsq[8];
    const int row = blockIdx.x;
    const int tid = threadIdx.x;
    const int wv  = tid >> 5, lane = tid & 31;

    f32x4 xv = *(const f32x4*)(x + (size_t)row * CD + tid * 4);
    float s = xv[0] + xv[1] + xv[2] + xv[3];
    float q = xv[0] * xv[0] + xv[1] * xv[1] + xv[2] * xv[2] + xv[3] * xv[3];
#pragma unroll
    for (int off = 16; off > 0; off >>= 1) {
        s += __shfl_xor(s, off);
        q += __shfl_xor(q, off);
    }
    if (lane == 0) { rsum[wv] = s; rsq[wv] = q; }
    __syncthreads();
    float ts = 0.0f, tq = 0.0f;
#pragma unroll
    for (int i = 0; i < 8; ++i) { ts += rsum[i]; tq += rsq[i]; }
    const float mean = ts * (1.0f / CD);
    const float var  = tq * (1.0f / CD) - mean * mean;
    const float rstd = rsqrtf(var + 1e-5f);
#pragma unroll
    for (int e = 0; e < 4; ++e) {
        const int c = tid * 4 + e;
        out[(size_t)row * CD + c] = (xv[e] - mean) * rstd * gamma[c] + beta[c];
    }
}

// ---------------------------------------------------------------------------
extern "C" void kernel_launch(void* const* d_in, const int* in_sizes, int n_in,
                              void* d_out, int out_size, void* d_ws, size_t ws_size,
                              hipStream_t stream) {
    const float* Q    = (const float*)d_in[0];
    const float* K    = (const float*)d_in[1];
    const float* V    = (const float*)d_in[2];
    const unsigned char* mask = (const unsigned char*)d_in[3]; // jnp bool: 1B
    const float* wq   = (const float*)d_in[4];
    const float* bq   = (const float*)d_in[5];
    const float* wk   = (const float*)d_in[6];
    const float* bk   = (const float*)d_in[7];
    const float* wv   = (const float*)d_in[8];
    const float* bv   = (const float*)d_in[9];
    const float* wo   = (const float*)d_in[10];
    const float* bo   = (const float*)d_in[11];
    const float* gamma= (const float*)d_in[12];
    const float* beta = (const float*)d_in[13];

    const size_t HE = (size_t)CB * CH * CS * CDK;   // 4,194,304 halves
    half_t*   qh   = (half_t*)d_ws;
    half_t*   khd  = qh  + HE;
    half_t*   vT   = khd + HE;
    half_t*   ctx  = vT  + HE;
    float*    xres = (float*)(ctx + HE);
    uint32_t* bits = (uint32_t*)(xres + (size_t)CB * CS * CD);

    mask_pack_kernel<<<CB * CS, 256, 0, stream>>>(mask, bits);

    dim3 gg(CD / 64, (CB * CS) / 256);   // (16, 16)
    gemm_kernel<float,  0><<<gg, 256, 0, stream>>>(Q, wq, bq, nullptr, qh,  nullptr);
    gemm_kernel<float,  0><<<gg, 256, 0, stream>>>(K, wk, bk, nullptr, khd, nullptr);
    gemm_kernel<float,  1><<<gg, 256, 0, stream>>>(V, wv, bv, nullptr, vT,  nullptr);

    flash_attn_kernel<<<dim3(CS / 128, CH, CB), 256, 0, stream>>>(qh, khd, vT, bits, ctx);

    gemm_kernel<half_t, 2><<<gg, 256, 0, stream>>>(ctx, wo, bo, Q, nullptr, xres);

    layernorm_kernel<<<CB * CS, 256, 0, stream>>>(xres, gamma, beta, (float*)d_out);
    (void)in_sizes; (void)n_in; (void)out_size; (void)ws_size;
}
